// MapEncoder_18957985644653
// MI455X (gfx1250) — compile-verified, matrix operand order checked by reading, not run on hardware
//
#include <hip/hip_runtime.h>
#include <hip/hip_bf16.h>

// ---------------------------------------------------------------------------
// MapEncoder for MI455X (gfx1250).  All heavy GEMMs go through
// v_wmma_f32_16x16x32_f16 (f16 inputs, f32 accumulate).  Activations are kept
// in f16 in workspace to halve HBM traffic (this workload is bound by the
// R x 256 intermediate tensors, not by math: ~160 GFLOP vs ~1.5 GB traffic).
// ---------------------------------------------------------------------------

typedef __attribute__((ext_vector_type(16))) _Float16 v16h;
typedef __attribute__((ext_vector_type(8)))  float    v8f;

union V16H {
    v16h   v;
    float4 q[2];   // two 16-byte halves of the 32-byte fragment
};

#define Bsz 16
#define Nn  1024
#define Mm  21
#define Dd  256
#define NPOLY (Bsz * Nn)        // 16384 polylines
#define RROWS (NPOLY * 20)      // 327680 segment rows
#define BN_EPS 1e-5f

// ---------------------------------------------------------------------------
// Generic WMMA GEMM:  out[M x N] = A[M x K] (f16, row-major) * Bt[N x K] (f16)
// Each wave computes a 16x64 tile (A fragment reused across 4 B tiles).
// Block = 256 threads = 8 waves -> 128 rows x 64 cols per block.
// grid = (rows/128, N/64).  rows % 128 == 0, K % 32 == 0, N % 64 == 0.
// Epilogue (all optional, runtime): +bias[n], +addrow[row/rpp][n] (pooled*W3b
// term), batchnorm, relu, mask-zero by row/rpp, output f16 or f32.
// ---------------------------------------------------------------------------
__global__ void gemm_wmma_k(const _Float16* __restrict__ A,
                            const _Float16* __restrict__ Bt,
                            int K, int N,
                            const float* __restrict__ bias,
                            const float* __restrict__ addrow,
                            const float* __restrict__ bn_g,
                            const float* __restrict__ bn_b,
                            const float* __restrict__ bn_m,
                            const float* __restrict__ bn_v,
                            const int*   __restrict__ mask,
                            int rowsPerPoly,
                            int doRelu,
                            _Float16* __restrict__ outH,
                            float*    __restrict__ outF)
{
    const int wave = threadIdx.x >> 5;
    const int lane = threadIdx.x & 31;
    const int half = lane >> 4;     // which 16-lane half of the wave
    const int lr   = lane & 15;

    const int rowBase = blockIdx.x * 128 + wave * 16;
    const int colBase = blockIdx.y * 64;

    // A fragment: row = rowBase+lr, K chunks at half*8 and 16+half*8.
    const _Float16* aPtr  = A  + (size_t)(rowBase + lr) * K + half * 8;
    const _Float16* bPtr0 = Bt + (size_t)(colBase +  0 + lr) * K + half * 8;
    const _Float16* bPtr1 = Bt + (size_t)(colBase + 16 + lr) * K + half * 8;
    const _Float16* bPtr2 = Bt + (size_t)(colBase + 32 + lr) * K + half * 8;
    const _Float16* bPtr3 = Bt + (size_t)(colBase + 48 + lr) * K + half * 8;

    v8f acc[4] = {v8f{}, v8f{}, v8f{}, v8f{}};

    for (int kk = 0; kk < K; kk += 32) {
        V16H a, b0, b1, b2, b3;
        a.q[0]  = *(const float4*)(aPtr  + kk);
        a.q[1]  = *(const float4*)(aPtr  + kk + 16);
        b0.q[0] = *(const float4*)(bPtr0 + kk);
        b0.q[1] = *(const float4*)(bPtr0 + kk + 16);
        b1.q[0] = *(const float4*)(bPtr1 + kk);
        b1.q[1] = *(const float4*)(bPtr1 + kk + 16);
        b2.q[0] = *(const float4*)(bPtr2 + kk);
        b2.q[1] = *(const float4*)(bPtr2 + kk + 16);
        b3.q[0] = *(const float4*)(bPtr3 + kk);
        b3.q[1] = *(const float4*)(bPtr3 + kk + 16);

        acc[0] = __builtin_amdgcn_wmma_f32_16x16x32_f16(false, a.v, false, b0.v,
                                                        (short)0, acc[0], false, false);
        acc[1] = __builtin_amdgcn_wmma_f32_16x16x32_f16(false, a.v, false, b1.v,
                                                        (short)0, acc[1], false, false);
        acc[2] = __builtin_amdgcn_wmma_f32_16x16x32_f16(false, a.v, false, b2.v,
                                                        (short)0, acc[2], false, false);
        acc[3] = __builtin_amdgcn_wmma_f32_16x16x32_f16(false, a.v, false, b3.v,
                                                        (short)0, acc[3], false, false);
    }

    // Epilogue.  D layout: element r of acc[t] -> row = half*8 + r, col = lr.
#pragma unroll
    for (int t = 0; t < 4; ++t) {
        const int n = colBase + t * 16 + lr;
        const float bia = bias ? bias[n] : 0.0f;
        float bnScale = 1.0f, bnShift = 0.0f;
        if (bn_g) {
            bnScale = bn_g[n] * rsqrtf(bn_v[n] + BN_EPS);
            bnShift = bn_b[n] - bn_m[n] * bnScale;
        }
#pragma unroll
        for (int r = 0; r < 8; ++r) {
            const int m = rowBase + half * 8 + r;
            float val = acc[t][r] + bia;
            if (addrow) val += addrow[(size_t)(m / rowsPerPoly) * N + n];
            if (bn_g)   val = val * bnScale + bnShift;
            if (doRelu) val = fmaxf(val, 0.0f);
            if (mask && !mask[m / rowsPerPoly]) val = 0.0f;
            const size_t oi = (size_t)m * N + n;
            if (outH) outH[oi] = (_Float16)val;
            else      outF[oi] = val;
        }
    }
}

// dst[n*K + k] = (f16) src[k*N + n]   (weight transpose + downconvert)
__global__ void trans_f16_k(const float* __restrict__ src,
                            _Float16* __restrict__ dst, int K, int N)
{
    int idx = blockIdx.x * blockDim.x + threadIdx.x;
    if (idx >= K * N) return;
    int n = idx / K;
    int k = idx - n * K;
    dst[idx] = (_Float16)src[(size_t)k * N + n];
}

// Stage 1: geometric features -> Linear(6,128) + BN + ReLU -> f16.
// One block (128 threads) per segment row.
__global__ void featurize_k(const float* __restrict__ tgt,   // (B,N,21,2)
                            const int*   __restrict__ mask,  // (B,N)
                            const float* __restrict__ W1, const float* __restrict__ b1,
                            const float* __restrict__ g1, const float* __restrict__ be1,
                            const float* __restrict__ m1, const float* __restrict__ v1,
                            _Float16* __restrict__ h1)        // (R,128)
{
    const int row  = blockIdx.x;         // 0..R-1
    const int poly = row / 20;
    const int p    = row - poly * 20;

    __shared__ float feat[6];
    if (threadIdx.x == 0) {
        const float* pts = tgt + (size_t)poly * (Mm * 2);
        const float x0 = pts[p * 2],     y0 = pts[p * 2 + 1];
        const float x1 = pts[p * 2 + 2], y1 = pts[p * 2 + 3];
        const float cx = pts[20],        cy = pts[21];        // point M/2 = 10
        const float pvx = x1 - x0, pvy = y1 - y0;
        const float vn  = sqrtf(pvx * pvx + pvy * pvy)
                        + (mask[poly] ? 1.0f : 0.0f) + 1e-6f;
        feat[0] = x0 - cx; feat[1] = y0 - cy;
        feat[2] = pvx;     feat[3] = pvy;
        feat[4] = pvx / vn; feat[5] = pvy / vn;
    }
    __syncthreads();

    const int d = threadIdx.x;           // 0..127
    float s = b1[d];
#pragma unroll
    for (int c = 0; c < 6; ++c) s += feat[c] * W1[c * 128 + d];
    s = (s - m1[d]) * rsqrtf(v1[d] + BN_EPS) * g1[d] + be1[d];
    s = fmaxf(s, 0.0f);
    h1[(size_t)row * 128 + d] = (_Float16)s;
}

// pooled[poly][d] = max over 20 rows of H (already mask-zeroed)
__global__ void poolmax_k(const _Float16* __restrict__ H,
                          _Float16* __restrict__ pooled, int nTot)
{
    int idx = blockIdx.x * blockDim.x + threadIdx.x;
    if (idx >= nTot) return;
    const int poly = idx >> 8, d = idx & 255;
    const _Float16* p = H + (size_t)poly * 20 * 256 + d;
    float mx = -INFINITY;
    for (int i = 0; i < 20; ++i) mx = fmaxf(mx, (float)p[i * 256]);
    pooled[idx] = (_Float16)mx;
}

// max-pool over 20 rows of G, + type_emb[label], mask -> map_feats (f32)
__global__ void finalreduce_k(const _Float16* __restrict__ G,
                              const float* __restrict__ type_emb,
                              const int*   __restrict__ label,
                              const int*   __restrict__ mask,
                              float* __restrict__ out)
{
    int idx = blockIdx.x * blockDim.x + threadIdx.x;
    if (idx >= NPOLY * Dd) return;
    const int poly = idx >> 8, d = idx & 255;
    const _Float16* p = G + (size_t)poly * 20 * 256 + d;
    float mx = -INFINITY;
    for (int i = 0; i < 20; ++i) mx = fmaxf(mx, (float)p[i * 256]);
    const float v = mx + type_emb[label[poly] * Dd + d];
    out[idx] = mask[poly] ? v : 0.0f;
}

// Sine positional embedding of polyline centers -> (NPOLY, 256) f16.
// Matches _gen_sineembed: out = concat([ey, ex]); for k in [0,128):
//   e[k] = (k odd ? cos : sin)(pos * 2*pi / 10000^{floor(k/2)/64})
__global__ void sine_k(const float* __restrict__ tgt,
                       const float* __restrict__ pc,
                       _Float16* __restrict__ sine)
{
    const int poly = blockIdx.x;
    const int d    = threadIdx.x;        // 0..255
    const float* pts = tgt + (size_t)poly * (Mm * 2);
    const float cx = pts[20], cy = pts[21];
    const float posx = (cx - pc[0]) / (pc[3] - pc[0]);
    const float posy = (cy - pc[1]) / (pc[4] - pc[1]);
    const float pos  = (d < 128) ? posy : posx;   // ey first, then ex
    const int   k    = d & 127;
    const float dimt = powf(10000.0f, (float)(k >> 1) * (1.0f / 64.0f));
    const float ph   = pos * 6.283185307179586f / dimt;
    const float v    = (k & 1) ? cosf(ph) : sinf(ph);
    sine[(size_t)poly * 256 + d] = (_Float16)v;
}

// third output: ~mask as float
__global__ void notmask_k(const int* __restrict__ mask, float* __restrict__ out)
{
    int i = blockIdx.x * blockDim.x + threadIdx.x;
    if (i < NPOLY) out[i] = mask[i] ? 0.0f : 1.0f;
}

// ---------------------------------------------------------------------------
extern "C" void kernel_launch(void* const* d_in, const int* in_sizes, int n_in,
                              void* d_out, int out_size, void* d_ws, size_t ws_size,
                              hipStream_t stream)
{
    (void)in_sizes; (void)n_in; (void)out_size; (void)ws_size;

    const float* tgt      = (const float*)d_in[0];
    const int*   label    = (const int*)  d_in[1];
    const int*   mask     = (const int*)  d_in[2];
    const float* W1       = (const float*)d_in[3];
    const float* b1       = (const float*)d_in[4];
    const float* g1       = (const float*)d_in[5];
    const float* be1      = (const float*)d_in[6];
    const float* m1       = (const float*)d_in[7];
    const float* v1       = (const float*)d_in[8];
    const float* W2       = (const float*)d_in[9];
    const float* b2       = (const float*)d_in[10];
    const float* W3       = (const float*)d_in[11];
    const float* b3       = (const float*)d_in[12];
    const float* g2       = (const float*)d_in[13];
    const float* be2      = (const float*)d_in[14];
    const float* m2       = (const float*)d_in[15];
    const float* v2       = (const float*)d_in[16];
    const float* W4       = (const float*)d_in[17];
    const float* b4       = (const float*)d_in[18];
    const float* type_emb = (const float*)d_in[19];
    const float* Wp1      = (const float*)d_in[20];
    const float* bp1      = (const float*)d_in[21];
    const float* Wp2      = (const float*)d_in[22];
    const float* bp2      = (const float*)d_in[23];
    const float* pc       = (const float*)d_in[24];

    float* outFeats = (float*)d_out;                       // (16384, 256)
    float* outPos   = outFeats + (size_t)NPOLY * Dd;       // (16384, 256)
    float* outMask  = outPos   + (size_t)NPOLY * Dd;       // (16384,)

    // ---- workspace layout (256B aligned regions) ----
    char* ws = (char*)d_ws;
    size_t off = 0;
    auto alloc = [&](size_t bytes) -> void* {
        off = (off + 255) & ~(size_t)255;
        void* p = ws + off;
        off += bytes;
        return p;
    };
    _Float16* W2t   = (_Float16*)alloc((size_t)256 * 128 * 2);
    _Float16* W3at  = (_Float16*)alloc((size_t)256 * 256 * 2);
    _Float16* W3bt  = (_Float16*)alloc((size_t)256 * 256 * 2);
    _Float16* W4t   = (_Float16*)alloc((size_t)256 * 256 * 2);
    _Float16* Wp1t  = (_Float16*)alloc((size_t)512 * 256 * 2);
    _Float16* Wp2t  = (_Float16*)alloc((size_t)256 * 512 * 2);
    _Float16* h1    = (_Float16*)alloc((size_t)RROWS * 128 * 2);
    _Float16* Hbuf  = (_Float16*)alloc((size_t)RROWS * 256 * 2); // H, later reused for G
    _Float16* H2buf = (_Float16*)alloc((size_t)RROWS * 256 * 2);
    _Float16* pooled= (_Float16*)alloc((size_t)NPOLY * 256 * 2);
    float*    P2    = (float*)   alloc((size_t)NPOLY * 256 * 4);
    _Float16* sineb = (_Float16*)alloc((size_t)NPOLY * 256 * 2);
    _Float16* peHid = (_Float16*)alloc((size_t)NPOLY * 512 * 2);

    // ---- 0. weight transpose + f16 convert ----
    trans_f16_k<<<(128 * 256 + 255) / 256, 256, 0, stream>>>(W2, W2t, 128, 256);
    trans_f16_k<<<(256 * 256 + 255) / 256, 256, 0, stream>>>(W3, W3at, 256, 256);
    trans_f16_k<<<(256 * 256 + 255) / 256, 256, 0, stream>>>(W3 + 256 * 256, W3bt, 256, 256);
    trans_f16_k<<<(256 * 256 + 255) / 256, 256, 0, stream>>>(W4, W4t, 256, 256);
    trans_f16_k<<<(256 * 512 + 255) / 256, 256, 0, stream>>>(Wp1, Wp1t, 256, 512);
    trans_f16_k<<<(512 * 256 + 255) / 256, 256, 0, stream>>>(Wp2, Wp2t, 512, 256);

    // ---- 1. features + Linear(6,128)+BN+ReLU ----
    featurize_k<<<RROWS, 128, 0, stream>>>(tgt, mask, W1, b1, g1, be1, m1, v1, h1);

    // ---- 2. H = (h1 @ W2 + b2), mask-zeroed, f16 ----
    dim3 gR(RROWS / 128, 256 / 64);
    gemm_wmma_k<<<gR, 256, 0, stream>>>(h1, W2t, 128, 256, b2,
                                        nullptr, nullptr, nullptr, nullptr, nullptr,
                                        mask, 20, 0, Hbuf, nullptr);

    // ---- 3. pooled = max_p H ----
    poolmax_k<<<(NPOLY * 256 + 255) / 256, 256, 0, stream>>>(Hbuf, pooled, NPOLY * 256);

    // ---- 4. P2 = pooled @ W3b  (f32) ----
    dim3 gP(NPOLY / 128, 256 / 64);
    gemm_wmma_k<<<gP, 256, 0, stream>>>(pooled, W3bt, 256, 256, nullptr,
                                        nullptr, nullptr, nullptr, nullptr, nullptr,
                                        nullptr, 1, 0, nullptr, P2);

    // ---- 5. H2 = relu(bn(H @ W3a + P2[poly] + b3)), f16 ----
    gemm_wmma_k<<<gR, 256, 0, stream>>>(Hbuf, W3at, 256, 256, b3,
                                        P2, g2, be2, m2, v2,
                                        nullptr, 20, 1, H2buf, nullptr);

    // ---- 6. G = (H2 @ W4 + b4), mask-zeroed, f16 (reuse Hbuf region) ----
    gemm_wmma_k<<<gR, 256, 0, stream>>>(H2buf, W4t, 256, 256, b4,
                                        nullptr, nullptr, nullptr, nullptr, nullptr,
                                        mask, 20, 0, Hbuf, nullptr);

    // ---- 7. map_feats = mask ? (max_p G + type_emb[label]) : 0 ----
    finalreduce_k<<<(NPOLY * 256 + 255) / 256, 256, 0, stream>>>(
        Hbuf, type_emb, label, mask, outFeats);

    // ---- 8. sine positional embedding ----
    sine_k<<<NPOLY, 256, 0, stream>>>(tgt, pc, sineb);

    // ---- 9. peHid = relu(sine @ Wp1 + bp1), f16 ----
    dim3 gS(NPOLY / 128, 512 / 64);
    gemm_wmma_k<<<gS, 256, 0, stream>>>(sineb, Wp1t, 256, 512, bp1,
                                        nullptr, nullptr, nullptr, nullptr, nullptr,
                                        nullptr, 1, 1, peHid, nullptr);

    // ---- 10. x_pos = mask ? (peHid @ Wp2 + bp2) : 0  (f32 -> d_out) ----
    gemm_wmma_k<<<gP, 256, 0, stream>>>(peHid, Wp2t, 512, 256, bp2,
                                        nullptr, nullptr, nullptr, nullptr, nullptr,
                                        mask, 1, 0, nullptr, outPos);

    // ---- 11. ~mask ----
    notmask_k<<<(NPOLY + 255) / 256, 256, 0, stream>>>(mask, outMask);
}
